// PerformerAttention_15530601742606
// MI455X (gfx1250) — matrix operand order repaired and checked by reading, not compiled, not run on hardware
//
#include <hip/hip_runtime.h>
#include <stdint.h>
#include <math.h>

typedef __bf16 bf16;
typedef __attribute__((ext_vector_type(16))) __bf16 bf16x16;
typedef __attribute__((ext_vector_type(8)))  float  floatx8;

// ---------------------------------------------------------------------------
// gfx1250 async global->LDS copy (ASYNCcnt-tracked) + wait
// ---------------------------------------------------------------------------
__device__ __forceinline__ void async_copy_b128(const bf16* g, const bf16* l) {
  asm volatile("global_load_async_to_lds_b128 %0, %1, off"
               :: "v"((uint32_t)(uintptr_t)l), "v"((uint64_t)(uintptr_t)g)
               : "memory");
}
__device__ __forceinline__ void async_wait0() {
  asm volatile("s_wait_asynccnt 0x0" ::: "memory");
}

__device__ __forceinline__ floatx8 wmma_bf16(bf16x16 a, bf16x16 b, floatx8 c) {
  return __builtin_amdgcn_wmma_f32_16x16x32_bf16(false, a, false, b, (short)0, c,
                                                 false, false);
}

// ---------------------------------------------------------------------------
// elementwise f32 -> bf16 (weights keep their natural [n][k] layout)
// ---------------------------------------------------------------------------
__global__ void __launch_bounds__(256) convert_bf16(const float* __restrict__ W,
                                                    bf16* __restrict__ O) {
  size_t i = ((size_t)blockIdx.x * 256 + threadIdx.x) * 4;
  float4 f = *(const float4*)(W + i);
  O[i + 0] = (bf16)f.x; O[i + 1] = (bf16)f.y;
  O[i + 2] = (bf16)f.z; O[i + 3] = (bf16)f.w;
}

// ---------------------------------------------------------------------------
// Ksum[b][d] = sum_seq Kt[d][b*Nn + seq]   (Kt is [D][B*Nn] row-major)
// ---------------------------------------------------------------------------
__global__ void __launch_bounds__(256) ksum_kernel(const bf16* __restrict__ Kt,
                                                   float* __restrict__ Ksum,
                                                   int ld, int Nn, int D) {
  int d = blockIdx.x * 256 + threadIdx.x;
  int b = blockIdx.y;
  const bf16* p = Kt + (size_t)d * ld + (size_t)b * Nn;
  float acc = 0.f;
  for (int n = 0; n < Nn; n += 8) {
    union { uint4 u; bf16 h[8]; } t;
    t.u = *(const uint4*)(p + n);
#pragma unroll
    for (int j = 0; j < 8; ++j) acc += (float)t.h[j];
  }
  Ksum[(size_t)b * D + d] = acc;
}

// ---------------------------------------------------------------------------
// z[b][n] = 1 / (Q[b][n][:] . Ksum[b][:] + 1e-6)
// ---------------------------------------------------------------------------
__global__ void __launch_bounds__(256) z_kernel(const bf16* __restrict__ Q,
                                                const float* __restrict__ Ksum,
                                                float* __restrict__ z,
                                                int Nn, int D) {
  int n = blockIdx.x * 256 + threadIdx.x;
  int b = blockIdx.y;
  const bf16*  q  = Q + ((size_t)b * Nn + n) * D;
  const float* ks = Ksum + (size_t)b * D;
  float acc = 0.f;
  for (int d = 0; d < D; d += 8) {
    union { uint4 u; bf16 h[8]; } t;
    t.u = *(const uint4*)(q + d);
#pragma unroll
    for (int j = 0; j < 8; ++j) acc += (float)t.h[j] * ks[d + j];
  }
  z[(size_t)b * Nn + n] = 1.0f / (acc + 1e-6f);
}

// ---------------------------------------------------------------------------
// WMMA bf16 GEMM:  C[M,N] = epi( A[M,K] @ Bcol^T )  where Bcol[n][k] = B[k][n]
//   Block tile 128x128, BLK_K=32, 256 thr = 8 waves, wave tile 32x64.
//   Double-buffered LDS, async global->LDS fills, one barrier per k-step.
//   AMODE: 0 = A f32 (cvt through regs)   1 = A bf16 (async fill)
//   EPI:   0 = phi(x+bias)  1 = x+bias  2 = raw  3 = x*z[row]  4 = x+bias->f32
//   CT:    store C transposed (Ct[n][m], vectorized b128 stores)
// ---------------------------------------------------------------------------
template <int AMODE, int EPI, bool CT>
__global__ void __launch_bounds__(256)
gemm_wmma(const void* __restrict__ Aptr, const bf16* __restrict__ Bcol,
          const float* __restrict__ bias, const float* __restrict__ zvec,
          void* __restrict__ Cptr, int M, int N, int Kdim, int lda, int ldb,
          long long sA, long long sB, long long sC, long long sZ) {
  __shared__ __align__(16) bf16 Als[2][128][40];
  __shared__ __align__(16) bf16 Bls[2][128][40];

  const int b     = blockIdx.z;
  const int tileM = blockIdx.y * 128;
  const int tileN = blockIdx.x * 128;
  const int tid   = threadIdx.x;
  const int lane  = tid & 31;
  const int wave  = tid >> 5;
  const int wm    = wave & 3;   // 4 M-strips of 32 rows
  const int wn    = wave >> 2;  // 2 N-halves of 64 cols
  const int hsel  = lane >> 4;
  const int l15   = lane & 15;

  const float* Af = (const float*)Aptr + (size_t)b * sA;
  const bf16*  Ab = (const bf16*)Aptr + (size_t)b * sA;
  const bf16*  Bb = Bcol + (size_t)b * sB;

  // per-thread fill coords: 128 rows x 32 k, 16 elems (=2 x b128) per thread
  const int fr = tid >> 1;
  const int fk = (tid & 1) * 16;

  floatx8 acc[2][4];
#pragma unroll
  for (int s = 0; s < 2; ++s)
#pragma unroll
    for (int nt = 0; nt < 4; ++nt)
#pragma unroll
      for (int j = 0; j < 8; ++j) acc[s][nt][j] = 0.0f;

  float4 fst[4];  // staged f32 A chunk (AMODE==0)

  auto fillA_async = [&](int buf, int k0) {
    const bf16* g = Ab + (size_t)(tileM + fr) * lda + k0 + fk;
    async_copy_b128(g,     &Als[buf][fr][fk]);
    async_copy_b128(g + 8, &Als[buf][fr][fk + 8]);
  };
  auto loadA_f32 = [&](int k0) {
    const float4* s = (const float4*)(Af + (size_t)(tileM + fr) * lda + k0 + fk);
    fst[0] = s[0]; fst[1] = s[1]; fst[2] = s[2]; fst[3] = s[3];
    if (k0 + 32 < Kdim)
      __builtin_prefetch(Af + (size_t)(tileM + fr) * lda + k0 + 32 + fk, 0, 1);
  };
  auto storeA_f32 = [&](int buf) {
    union { uint4 u[2]; bf16 h[16]; } t;
#pragma unroll
    for (int i = 0; i < 4; ++i) {
      t.h[i * 4 + 0] = (bf16)fst[i].x; t.h[i * 4 + 1] = (bf16)fst[i].y;
      t.h[i * 4 + 2] = (bf16)fst[i].z; t.h[i * 4 + 3] = (bf16)fst[i].w;
    }
    uint4* d = (uint4*)&Als[buf][fr][fk];
    d[0] = t.u[0]; d[1] = t.u[1];
  };
  auto fillB_async = [&](int buf, int k0) {
    const bf16* g = Bb + (size_t)(tileN + fr) * ldb + k0 + fk;
    async_copy_b128(g,     &Bls[buf][fr][fk]);
    async_copy_b128(g + 8, &Bls[buf][fr][fk + 8]);
  };
  auto compute = [&](int buf) {
    // A fragment layout (16-bit 16x32): lanes 0-15 K=0..7,16..23 ; 16-31 +8
    bf16x16 af[2];
#pragma unroll
    for (int s = 0; s < 2; ++s) {
      union { uint4 u[2]; bf16x16 v; } ta;
      const uint4* pa = (const uint4*)&Als[buf][wm * 32 + s * 16 + l15][hsel * 8];
      ta.u[0] = pa[0];   // K kbase .. kbase+7
      ta.u[1] = pa[2];   // K kbase+16 .. kbase+23
      af[s] = ta.v;
    }
#pragma unroll
    for (int nt = 0; nt < 4; ++nt) {
      union { uint4 u[2]; bf16x16 v; } tb;
      const uint4* pb = (const uint4*)&Bls[buf][wn * 64 + nt * 16 + l15][hsel * 8];
      tb.u[0] = pb[0]; tb.u[1] = pb[2];
      acc[0][nt] = wmma_bf16(af[0], tb.v, acc[0][nt]);
      acc[1][nt] = wmma_bf16(af[1], tb.v, acc[1][nt]);
    }
  };

  // ---- prologue: fill buffer 0 ----
  if (AMODE == 0) { loadA_f32(0); storeA_f32(0); }
  else            { fillA_async(0, 0); }
  fillB_async(0, 0);
  async_wait0();
  __syncthreads();

  // ---- pipelined main loop: one barrier per k-step ----
  for (int k0 = 0; k0 < Kdim; k0 += 32) {
    const int  p   = (k0 >> 5) & 1;
    const bool nxt = (k0 + 32) < Kdim;
    if (nxt) {
      if (AMODE == 0) loadA_f32(k0 + 32);
      else            fillA_async(1 - p, k0 + 32);
      fillB_async(1 - p, k0 + 32);
    }
    compute(p);                    // overlaps in-flight async/global loads
    if (nxt) {
      if (AMODE == 0) storeA_f32(1 - p);
      async_wait0();
      __syncthreads();
    }
  }

  // ---- epilogue: C/D layout vgpr r -> M = r + hsel*8, N = l15 ----
#pragma unroll
  for (int s = 0; s < 2; ++s) {
#pragma unroll
    for (int nt = 0; nt < 4; ++nt) {
      const int rowb = tileM + wm * 32 + s * 16 + hsel * 8;
      const int col  = tileN + wn * 64 + nt * 16 + l15;
      float v[8];
#pragma unroll
      for (int r = 0; r < 8; ++r) {
        float x = acc[s][nt][r];
        if (EPI == 0 || EPI == 1 || EPI == 4) x += bias[col];
        if (EPI == 0) x = __expf(-0.5f * x * x);
        if (EPI == 3) x *= zvec[(size_t)b * sZ + rowb + r];
        v[r] = x;
      }
      if (CT) {  // transposed store: 8 consecutive rows -> one b128
        union { uint4 u; bf16 h[8]; } t;
#pragma unroll
        for (int r = 0; r < 8; ++r) t.h[r] = (bf16)v[r];
        *(uint4*)((bf16*)Cptr + (size_t)b * sC + (size_t)col * M + rowb) = t.u;
      } else if (EPI == 4) {
        float* C = (float*)Cptr + (size_t)b * sC;
#pragma unroll
        for (int r = 0; r < 8; ++r) C[(size_t)(rowb + r) * N + col] = v[r];
      } else {
        bf16* C = (bf16*)Cptr + (size_t)b * sC;
#pragma unroll
        for (int r = 0; r < 8; ++r) C[(size_t)(rowb + r) * N + col] = (bf16)v[r];
      }
    }
  }
}

// ---------------------------------------------------------------------------
extern "C" void kernel_launch(void* const* d_in, const int* in_sizes, int n_in,
                              void* d_out, int out_size, void* d_ws, size_t ws_size,
                              hipStream_t stream) {
  (void)in_sizes; (void)n_in; (void)out_size; (void)ws_size;
  constexpr int D  = 1024;
  constexpr int B  = 4;
  constexpr int Nn = 4096;
  constexpr int BN = B * Nn;                    // 16384
  constexpr size_t BND = (size_t)BN * D;        // 16,777,216

  const float* q  = (const float*)d_in[0];
  const float* k  = (const float*)d_in[1];
  const float* v  = (const float*)d_in[2];
  const float* Wq = (const float*)d_in[3];
  const float* bq = (const float*)d_in[4];
  const float* Wk = (const float*)d_in[5];
  const float* bk = (const float*)d_in[6];
  const float* Wv = (const float*)d_in[7];
  const float* bv = (const float*)d_in[8];
  const float* Wo = (const float*)d_in[9];
  const float* bo = (const float*)d_in[10];

  // ---- workspace carve ----
  char* ws = (char*)d_ws;
  size_t o = 0;
  bf16* Wqb = (bf16*)(ws + o); o += (size_t)D * D * 2;   // W[n][k] bf16
  bf16* Wkb = (bf16*)(ws + o); o += (size_t)D * D * 2;
  bf16* Wvb = (bf16*)(ws + o); o += (size_t)D * D * 2;
  bf16* Wob = (bf16*)(ws + o); o += (size_t)D * D * 2;
  bf16* Qb  = (bf16*)(ws + o); o += BND * 2;             // [B*Nn][D]
  bf16* Ktb = (bf16*)(ws + o); o += BND * 2;             // [D][B*Nn]
  bf16* Vtb = (bf16*)(ws + o); o += BND * 2;             // [D][B*Nn]
  bf16* Sb  = (bf16*)(ws + o); o += BND * 2;             // [B*Nn][D]
  bf16* KVt = (bf16*)(ws + o); o += (size_t)B * D * D * 2; // [b][n][m]
  float* Ksum = (float*)(ws + o); o += (size_t)B * D * 4;
  float* Zv   = (float*)(ws + o); o += (size_t)B * Nn * 4;

  dim3 blk(256);

  // 1) weights -> bf16 (natural layout == Bcol layout)
  convert_bf16<<<dim3((D * D) / 1024), blk, 0, stream>>>(Wq, Wqb);
  convert_bf16<<<dim3((D * D) / 1024), blk, 0, stream>>>(Wk, Wkb);
  convert_bf16<<<dim3((D * D) / 1024), blk, 0, stream>>>(Wv, Wvb);
  convert_bf16<<<dim3((D * D) / 1024), blk, 0, stream>>>(Wo, Wob);

  // 2) projections (M = 16384, N = 1024, K = 1024)
  dim3 gproj(D / 128, BN / 128, 1);
  gemm_wmma<0, 0, false><<<gproj, blk, 0, stream>>>(   // Q = phi(q Wq^T + bq)
      q, Wqb, bq, nullptr, Qb, BN, D, D, D, D, 0, 0, 0, 0);
  gemm_wmma<0, 0, true><<<gproj, blk, 0, stream>>>(    // Kt = phi(k Wk^T + bk)^T
      k, Wkb, bk, nullptr, Ktb, BN, D, D, D, D, 0, 0, 0, 0);
  gemm_wmma<0, 1, true><<<gproj, blk, 0, stream>>>(    // Vt = (v Wv^T + bv)^T
      v, Wvb, bv, nullptr, Vtb, BN, D, D, D, D, 0, 0, 0, 0);

  // 3) Ksum over sequence (rows of Kt)
  ksum_kernel<<<dim3(D / 256, B), blk, 0, stream>>>(Ktb, Ksum, BN, Nn, D);

  // 4) KVt[b] = (K[b]^T V[b])^T : A = Kt rows, Bcol = Vt rows
  dim3 gkv(D / 128, D / 128, B);
  gemm_wmma<1, 2, true><<<gkv, blk, 0, stream>>>(
      Ktb, Vtb, nullptr, nullptr, KVt, D, D, Nn, BN, BN,
      (long long)Nn, (long long)Nn, (long long)D * D, 0);

  // 5) row normalizers
  z_kernel<<<dim3(Nn / 256, B), blk, 0, stream>>>(Qb, Ksum, Zv, Nn, D);

  // 6) S[b] = (Q[b] @ KV[b]) * z : Bcol = KVt
  dim3 gs(D / 128, Nn / 128, B);
  gemm_wmma<1, 3, false><<<gs, blk, 0, stream>>>(
      Qb, KVt, nullptr, Zv, Sb, Nn, D, D, D, D,
      (long long)Nn * D, (long long)D * D, (long long)Nn * D, (long long)Nn);

  // 7) out = S @ Wo^T + bo -> f32
  gemm_wmma<1, 4, false><<<gproj, blk, 0, stream>>>(
      Sb, Wob, bo, nullptr, d_out, BN, D, D, D, D, 0, 0, 0, 0);
}
